// MultiHeadAttn_36258113913399
// MI455X (gfx1250) — matrix operand order repaired and checked by reading, not compiled, not run on hardware
//
#include <hip/hip_runtime.h>
#include <hip/hip_bf16.h>
#include <math.h>

typedef __bf16 bf16_t;
typedef __attribute__((ext_vector_type(16))) __bf16 v16bf;
typedef __attribute__((ext_vector_type(8)))  float  v8f;
typedef int v4i_async __attribute__((vector_size(16)));   // builtin's pointee type

constexpr int Bn = 4;
constexpr int Sn = 2048;
constexpr int Dm = 1024;          // d_model (N and K of every GEMM here)
constexpr int Hn = 16;
constexpr int DK = 64;
constexpr int Mrows = Bn * Sn;    // 8192
constexpr float ATTN_SCALE = 0.125f;   // 1/sqrt(64)
constexpr float NEG_BIG = -1e9f;

#define AS_GLOBAL __attribute__((address_space(1)))
#define AS_SHARED __attribute__((address_space(3)))

#if defined(__HIP_DEVICE_COMPILE__) && __has_builtin(__builtin_amdgcn_global_load_async_to_lds_b128)
#define HAVE_ASYNC_LDS 1
#else
#define HAVE_ASYNC_LDS 0
#endif

#if defined(__HIP_DEVICE_COMPILE__) && __has_builtin(__builtin_amdgcn_s_wait_asynccnt)
#define WAIT_ASYNC(n) __builtin_amdgcn_s_wait_asynccnt(n)
#elif HAVE_ASYNC_LDS
#define WAIT_ASYNC(n) asm volatile("s_wait_asynccnt %0" ::"i"(n) : "memory")
#else
#define WAIT_ASYNC(n) ((void)0)
#endif

// ---------------------------------------------------------------------------
// helpers
// ---------------------------------------------------------------------------
__device__ inline unsigned pack_bf16x2(float a, float b) {
  union { __bf16 h[2]; unsigned u; } t;
  t.h[0] = (__bf16)a;
  t.h[1] = (__bf16)b;
  return t.u;
}

union FragU {
  uint4 q[2];
  v16bf v;
};

// Load a v16bf fragment from two 16-byte chunks (A frags: chunk1 = chunk0+16
// elements; B frags: chunk1 = chunk0+16 elements as well).
__device__ inline v16bf load_frag_pair(const bf16_t* p0, const bf16_t* p1) {
  FragU f;
  f.q[0] = *(const uint4*)p0;
  f.q[1] = *(const uint4*)p1;
  return f.v;
}

#if HAVE_ASYNC_LDS
__device__ inline void async_cp_b128(const void* gsrc, void* ldst) {
  __builtin_amdgcn_global_load_async_to_lds_b128(
      (AS_GLOBAL v4i_async*)(void*)gsrc, (AS_SHARED v4i_async*)ldst, 0, 0);
}
#endif

// ---------------------------------------------------------------------------
// GEMM: C[M,N] = A[M,K] * W[N,K]^T + bias[N]
//   mode 0: f32 out, row-major [M,N]
//   mode 1: bf16 out, split-head [B,H,S,dk]
//   mode 2: bf16 out, transposed split-head [B,H,dk,S]   (for V^T)
// M = 8192, N = K = 1024 (hardcoded).
// ---------------------------------------------------------------------------
__global__ __launch_bounds__(256) void gemm_bf16_wmma(
    const float* __restrict__ A, const float* __restrict__ W,
    const float* __restrict__ bias, void* __restrict__ out, int mode) {
  __shared__ bf16_t lA[128 * 40];   // 128 rows x 32 K (bf16), stride 40
  __shared__ bf16_t lB[128 * 40];

  const int tid  = threadIdx.x;
  const int wave = tid >> 5;
  const int lane = tid & 31;
  const int lh   = lane >> 4;      // lane half (K-split select)
  const int lm   = lane & 15;
  const int waveM = wave >> 2;     // 0..1 -> 64 rows each
  const int waveN = wave & 3;      // 0..3 -> 32 cols each
  const int blockN = blockIdx.x;   // N/128 = 8
  const int blockM = blockIdx.y;   // M/128 = 64

  const int ldRow = tid >> 3;      // 0..31 rows per pass
  const int ldC4  = tid & 7;       // float4 index within 32-wide K slab

  v8f acc[4][2];
#pragma unroll
  for (int m = 0; m < 4; ++m)
#pragma unroll
    for (int n = 0; n < 2; ++n) acc[m][n] = (v8f){0, 0, 0, 0, 0, 0, 0, 0};

  for (int k0 = 0; k0 < Dm; k0 += 32) {
    __syncthreads();
#pragma unroll
    for (int i = 0; i < 4; ++i) {
      const int r = ldRow + i * 32;
      const float4 av =
          *(const float4*)(A + (size_t)(blockM * 128 + r) * Dm + k0 + ldC4 * 4);
      *(uint2*)(&lA[r * 40 + ldC4 * 4]) =
          make_uint2(pack_bf16x2(av.x, av.y), pack_bf16x2(av.z, av.w));
      const float4 wv =
          *(const float4*)(W + (size_t)(blockN * 128 + r) * Dm + k0 + ldC4 * 4);
      *(uint2*)(&lB[r * 40 + ldC4 * 4]) =
          make_uint2(pack_bf16x2(wv.x, wv.y), pack_bf16x2(wv.z, wv.w));
    }
    __syncthreads();

    v16bf af[4], bfr[2];
#pragma unroll
    for (int m = 0; m < 4; ++m) {
      const bf16_t* pr = lA + (waveM * 64 + m * 16 + lm) * 40 + lh * 8;
      af[m] = load_frag_pair(pr, pr + 16);
    }
#pragma unroll
    for (int n = 0; n < 2; ++n) {
      const bf16_t* pr = lB + (waveN * 32 + n * 16 + lm) * 40 + lh * 8;
      bfr[n] = load_frag_pair(pr, pr + 16);
    }
#pragma unroll
    for (int m = 0; m < 4; ++m)
#pragma unroll
      for (int n = 0; n < 2; ++n)
        acc[m][n] = __builtin_amdgcn_wmma_f32_16x16x32_bf16(
            false, af[m], false, bfr[n], (short)0, acc[m][n], false, false);
  }

  // epilogue
#pragma unroll
  for (int n = 0; n < 2; ++n) {
    const int gn = blockN * 128 + waveN * 32 + n * 16 + lm;
    const float bv = bias[gn];
#pragma unroll
    for (int m = 0; m < 4; ++m) {
#pragma unroll
      for (int r = 0; r < 8; ++r) {
        const int gm = blockM * 128 + waveM * 64 + m * 16 + lh * 8 + r;
        const float v = acc[m][n][r] + bv;
        if (mode == 0) {
          ((float*)out)[(size_t)gm * Dm + gn] = v;
        } else {
          const int bb = gm >> 11, ss = gm & (Sn - 1);
          const int hh = gn >> 6, dd = gn & (DK - 1);
          bf16_t* o = (bf16_t*)out;
          if (mode == 1)
            o[(((size_t)(bb * Hn + hh)) * Sn + ss) * DK + dd] = (bf16_t)v;
          else
            o[(((size_t)(bb * Hn + hh)) * DK + dd) * Sn + ss] = (bf16_t)v;
        }
      }
    }
  }
}

// ---------------------------------------------------------------------------
// Flash attention: one workgroup per (b, h, 64-row q block); 4 waves x 16 rows.
// Q,K: bf16 [B,H,S,dk]; Vt: bf16 [B,H,dk,S]; out: f32 [B,S,D] (heads merged).
// K/V tiles staged in LDS once per workgroup via double-buffered async copies.
// ---------------------------------------------------------------------------
__global__ __launch_bounds__(128) void flash_attn_wmma(
    const bf16_t* __restrict__ Q, const bf16_t* __restrict__ K,
    const bf16_t* __restrict__ Vt, const int* __restrict__ mask,
    float* __restrict__ attnOut) {
  __shared__ bf16_t lK[2][64 * 72];   // K tile: 64 kv rows x 64 dk, stride 72
  __shared__ bf16_t lV[2][64 * 72];   // V^T tile: 64 dk rows x 64 kv, stride 72
  __shared__ bf16_t lP[4][16 * 72];   // per-wave 16x64 P tile, stride 72

  const int tid  = threadIdx.x;
  const int wave = tid >> 5;
  const int lane = tid & 31;
  const int lh   = lane >> 4, lm = lane & 15;

  const int qblk = blockIdx.x;   // S/64 = 32
  const int h    = blockIdx.y;   // 16
  const int b    = blockIdx.z;   // 4

  const int q0 = qblk * 64 + wave * 16;
  const bf16_t* Qb = Q + ((size_t)(b * Hn + h) * Sn) * DK;
  const bf16_t* Kb = K + ((size_t)(b * Hn + h) * Sn) * DK;
  const bf16_t* Vb = Vt + ((size_t)(b * Hn + h) * DK) * Sn;
  const int* Mb = mask + (size_t)b * Sn * Sn;

  // tile-copy assignment: thread copies 64 bytes of K and 64 bytes of V
  const int cpRow = tid >> 1;          // 0..63
  const int cpCh  = (tid & 1) * 32;    // bf16 offset within row

  auto issue_tile = [&](int j, int buf) {
    const bf16_t* gK = Kb + (size_t)(j * 64 + cpRow) * DK + cpCh;
    bf16_t* sK = &lK[buf][cpRow * 72 + cpCh];
    const bf16_t* gV = Vb + (size_t)cpRow * Sn + j * 64 + cpCh;
    bf16_t* sV = &lV[buf][cpRow * 72 + cpCh];
#if HAVE_ASYNC_LDS
#pragma unroll
    for (int o = 0; o < 32; o += 8) {   // 4 x b128 each for K and V
      async_cp_b128(gK + o, sK + o);
      async_cp_b128(gV + o, sV + o);
    }
#else
#pragma unroll
    for (int o = 0; o < 32; o += 8) {
      *(uint4*)(sK + o) = *(const uint4*)(gK + o);
      *(uint4*)(sV + o) = *(const uint4*)(gV + o);
    }
#endif
  };

  // Q A-fragments held in registers for the whole KV loop
  v16bf aQ[2];
#pragma unroll
  for (int kk = 0; kk < 2; ++kk) {
    const bf16_t* pr = Qb + (size_t)(q0 + lm) * DK + kk * 32 + lh * 8;
    aQ[kk] = load_frag_pair(pr, pr + 16);
  }

  float mrow[8], lrow[8];
#pragma unroll
  for (int r = 0; r < 8; ++r) { mrow[r] = -INFINITY; lrow[r] = 0.f; }
  v8f accO[4];
#pragma unroll
  for (int n = 0; n < 4; ++n) accO[n] = (v8f){0, 0, 0, 0, 0, 0, 0, 0};

  issue_tile(0, 0);
  int cur = 0;

  for (int j = 0; j < Sn / 64; ++j) {
    if (j + 1 < Sn / 64) {
      issue_tile(j + 1, cur ^ 1);   // next tile in flight during compute
      WAIT_ASYNC(8);                // retire current tile's 8 async ops
      __builtin_prefetch(Mb + (size_t)(q0 + lh * 8) * Sn + (j + 1) * 64 + lm,
                         0, 1);
    } else {
      WAIT_ASYNC(0);
    }
    __syncthreads();                // K/V tile visible to all waves

    // S = Q * K^T for this 16x64 stripe (B frags from LDS, conflict-free)
    v8f sAcc[4];
#pragma unroll
    for (int n = 0; n < 4; ++n) sAcc[n] = (v8f){0, 0, 0, 0, 0, 0, 0, 0};
#pragma unroll
    for (int n = 0; n < 4; ++n) {
#pragma unroll
      for (int kk = 0; kk < 2; ++kk) {
        const bf16_t* pr = &lK[cur][(n * 16 + lm) * 72 + kk * 32 + lh * 8];
        const v16bf bK = load_frag_pair(pr, pr + 16);
        sAcc[n] = __builtin_amdgcn_wmma_f32_16x16x32_bf16(
            false, aQ[kk], false, bK, (short)0, sAcc[n], false, false);
      }
    }

    // mask + scale; per-row running max
    float sv[4][8], rm[8];
#pragma unroll
    for (int r = 0; r < 8; ++r) rm[r] = -1e30f;
#pragma unroll
    for (int n = 0; n < 4; ++n) {
      const int kcol = j * 64 + n * 16 + lm;
#pragma unroll
      for (int r = 0; r < 8; ++r) {
        const int qrow = q0 + lh * 8 + r;
        const int mv = Mb[(size_t)qrow * Sn + kcol];
        const float s = (mv != 0) ? sAcc[n][r] * ATTN_SCALE : NEG_BIG;
        sv[n][r] = s;
        rm[r] = fmaxf(rm[r], s);
      }
    }
    // row-max across the 16 lanes sharing a row (lane halves hold rows r / r+8)
#pragma unroll
    for (int r = 0; r < 8; ++r)
#pragma unroll
      for (int off = 1; off < 16; off <<= 1)
        rm[r] = fmaxf(rm[r], __shfl_xor(rm[r], off, 32));

    float alpha[8];
#pragma unroll
    for (int r = 0; r < 8; ++r) {
      const float mnew = fmaxf(mrow[r], rm[r]);
      alpha[r] = __expf(mrow[r] - mnew);
      mrow[r] = mnew;
    }

    // P = exp(s - m): accumulate row sums and stash bf16 P in LDS (C->A reshape)
    float ps[8];
#pragma unroll
    for (int r = 0; r < 8; ++r) ps[r] = 0.f;
#pragma unroll
    for (int n = 0; n < 4; ++n) {
#pragma unroll
      for (int r = 0; r < 8; ++r) {
        const float p = __expf(sv[n][r] - mrow[r]);
        ps[r] += p;
        lP[wave][(lh * 8 + r) * 72 + n * 16 + lm] = (bf16_t)p;
      }
    }
#pragma unroll
    for (int r = 0; r < 8; ++r) {
#pragma unroll
      for (int off = 1; off < 16; off <<= 1)
        ps[r] += __shfl_xor(ps[r], off, 32);
      lrow[r] = lrow[r] * alpha[r] + ps[r];
    }
#pragma unroll
    for (int n = 0; n < 4; ++n)
#pragma unroll
      for (int r = 0; r < 8; ++r) accO[n][r] *= alpha[r];

    __syncthreads();
    v16bf aP[2];
#pragma unroll
    for (int kk = 0; kk < 2; ++kk) {
      const bf16_t* pr = &lP[wave][lm * 72 + kk * 32 + lh * 8];
      aP[kk] = load_frag_pair(pr, pr + 16);
    }
    // O += P * V   (V^T rows are d, contiguous in kv -> clean B fragments)
#pragma unroll
    for (int n = 0; n < 4; ++n) {
#pragma unroll
      for (int kk = 0; kk < 2; ++kk) {
        const bf16_t* pr = &lV[cur][(n * 16 + lm) * 72 + kk * 32 + lh * 8];
        const v16bf bV = load_frag_pair(pr, pr + 16);
        accO[n] = __builtin_amdgcn_wmma_f32_16x16x32_bf16(
            false, aP[kk], false, bV, (short)0, accO[n], false, false);
      }
    }
    __syncthreads();   // buffers + lP fully consumed before next overwrite
    cur ^= 1;
  }

  // normalize and write merged-head f32 output [B,S,D]
#pragma unroll
  for (int n = 0; n < 4; ++n) {
    const int d = h * DK + n * 16 + lm;
#pragma unroll
    for (int r = 0; r < 8; ++r) {
      const int qrow = q0 + lh * 8 + r;
      attnOut[(size_t)(b * Sn + qrow) * Dm + d] = accO[n][r] / lrow[r];
    }
  }
}

// ---------------------------------------------------------------------------
// launch
// ---------------------------------------------------------------------------
extern "C" void kernel_launch(void* const* d_in, const int* in_sizes, int n_in,
                              void* d_out, int out_size, void* d_ws,
                              size_t ws_size, hipStream_t stream) {
  (void)in_sizes; (void)n_in; (void)out_size; (void)ws_size;
  const float* x  = (const float*)d_in[0];
  const float* y  = (const float*)d_in[1];
  const int*  msk = (const int*)d_in[2];
  const float* Wq = (const float*)d_in[3];
  const float* bq = (const float*)d_in[4];
  const float* Wk = (const float*)d_in[5];
  const float* bk = (const float*)d_in[6];
  const float* Wv = (const float*)d_in[7];
  const float* bv = (const float*)d_in[8];
  const float* Wo = (const float*)d_in[9];
  const float* bo = (const float*)d_in[10];

  const size_t elems = (size_t)Mrows * Dm;           // 8Mi elements
  bf16_t* qb  = (bf16_t*)d_ws;                       // 16.8 MB
  bf16_t* kb  = qb + elems;                          // 16.8 MB
  bf16_t* vtb = kb + elems;                          // 16.8 MB (V^T)
  float* attnb = (float*)(vtb + elems);              // 33.5 MB

  dim3 gGrid(Dm / 128, Mrows / 128);                 // 8 x 64
  gemm_bf16_wmma<<<gGrid, 256, 0, stream>>>(x, Wq, bq, qb, 1);
  gemm_bf16_wmma<<<gGrid, 256, 0, stream>>>(y, Wk, bk, kb, 1);
  gemm_bf16_wmma<<<gGrid, 256, 0, stream>>>(y, Wv, bv, vtb, 2);
  flash_attn_wmma<<<dim3(Sn / 64, Hn, Bn), 128, 0, stream>>>(qb, kb, vtb, msk,
                                                             attnb);
  gemm_bf16_wmma<<<gGrid, 256, 0, stream>>>(attnb, Wo, bo, d_out, 0);
}
